// Self_mask_Spectral_MSA_35424890257927
// MI455X (gfx1250) — matrix operand order repaired and checked by reading, not compiled, not run on hardware
//
#include <hip/hip_runtime.h>
#include <math.h>

typedef __attribute__((ext_vector_type(16))) _Float16 v16h;
typedef __attribute__((ext_vector_type(8)))  _Float16 v8h;
typedef __attribute__((ext_vector_type(4)))  _Float16 v4h;
typedef __attribute__((ext_vector_type(8)))  float    v8f;

#define Bb 4
#define HH 256
#define WW 256
#define CC 32
#define NN (HH * WW)

__device__ inline v8f wmma16(v16h a, v16h b, v8f c) {
  return __builtin_amdgcn_wmma_f32_16x16x32_f16(false, a, false, b, (short)0, c,
                                                false, false);
}

// A fragment from row-major 16x32 f16 tile (row stride 32 halves, 16B aligned):
// lane holds row m = lane&15; halves j -> K = j + 8*((j>=8)+(lane>=16)):
// two contiguous 16B runs -> 2x ds_load_b128.
__device__ inline v16h frag_A(const _Float16* tile, int lane) {
  int m = lane & 15, hi = (lane >> 4) & 1;
  const _Float16* r = tile + m * 32;
  v8h lo = *(const v8h*)(r + 8 * hi);
  v8h hh = *(const v8h*)(r + 16 + 8 * hi);
  return __builtin_shufflevector(lo, hh, 0, 1, 2, 3, 4, 5, 6, 7, 8, 9, 10, 11,
                                 12, 13, 14, 15);
}

// B fragment from TRANSPOSED storage BT[n][k] (n-major, 32 halves per n):
// lane holds col n = lane&15; halves j -> K = j + 8*((j>=8)+(lane>=16)):
// two contiguous 16B runs, matching frag_A.
__device__ inline v16h frag_B(const _Float16* tileT, int lane) {
  int n = lane & 15, hi = (lane >> 4) & 1;
  const _Float16* r = tileT + n * 32;
  v8h lo = *(const v8h*)(r + 8 * hi);
  v8h hh = *(const v8h*)(r + 16 + 8 * hi);
  return __builtin_shufflevector(lo, hh, 0, 1, 2, 3, 4, 5, 6, 7, 8, 9, 10, 11,
                                 12, 13, 14, 15);
}

__device__ inline float gelu_exact(float x) {
  return 0.5f * x * (1.0f + erff(x * 0.70710678118654752440f));
}

// ---------- channel attention: spatial mean of x_hsi per (b,c) ----------
__global__ __launch_bounds__(256) void k_chan_avg(const float* __restrict__ xhsi,
                                                  float* __restrict__ avg) {
  __shared__ float part[32][32];
  int b = blockIdx.x >> 6;    // 64 blocks per batch, 1024 positions each
  int blk = blockIdx.x & 63;
  int t = threadIdx.x;
  int c4 = t & 7, g = t >> 3;  // 32 position groups x 8 float4 lanes
  const float4* base = (const float4*)(xhsi + (size_t)b * NN * CC);
  float4 s = {0.f, 0.f, 0.f, 0.f};
  for (int i = 0; i < 32; ++i) {
    int p = blk * 1024 + g * 32 + i;
    float4 f = base[p * 8 + c4];
    s.x += f.x; s.y += f.y; s.z += f.z; s.w += f.w;
  }
  part[g][c4 * 4 + 0] = s.x;
  part[g][c4 * 4 + 1] = s.y;
  part[g][c4 * 4 + 2] = s.z;
  part[g][c4 * 4 + 3] = s.w;
  __syncthreads();
  if (t < 32) {
    float tot = 0.f;
    for (int g2 = 0; g2 < 32; ++g2) tot += part[g2][t];
    atomicAdd(&avg[b * CC + t], tot);
  }
}

__global__ void k_chan_mask(const float* __restrict__ avg, const float* __restrict__ fc1,
                            const float* __restrict__ fc2, float* __restrict__ cm,
                            float* __restrict__ out_cm) {
  __shared__ float a[4][32], y1[4][32];
  int t = threadIdx.x;  // 128 threads
  int b = t >> 5, c = t & 31;
  a[b][c] = avg[t] * (1.0f / (float)NN);
  __syncthreads();
  float s = 0.f;
  for (int i = 0; i < 32; ++i) s += fc1[c * 32 + i] * a[b][i];
  y1[b][c] = fmaxf(s, 0.f);
  __syncthreads();
  s = 0.f;
  for (int i = 0; i < 32; ++i) s += fc2[c * 32 + i] * y1[b][i];
  float m = 1.f / (1.f + expf(-s));
  cm[t] = m;
  out_cm[t] = m;
}

// ---------- collapse conv(CxC) -> 1x1(Cx1) into effective Cx1 kernels ----------
__global__ void k_weff(const float* __restrict__ c1a, const float* __restrict__ c1b,
                       const float* __restrict__ c2a, const float* __restrict__ c2b,
                       float* __restrict__ weff1, float* __restrict__ weff2) {
  int t = threadIdx.x;
  for (int idx = t; idx < 32 * 9; idx += blockDim.x) {
    int i = idx / 9, tap = idx % 9;
    float s = 0.f;
    for (int o = 0; o < 32; ++o) s += c1b[o] * c1a[(o * 32 + i) * 9 + tap];
    weff1[tap * 32 + i] = s;
  }
  for (int idx = t; idx < 32 * 49; idx += blockDim.x) {
    int i = idx / 49, tap = idx % 49;
    float s = 0.f;
    for (int o = 0; o < 32; ++o) s += c2b[o] * c2a[(o * 32 + i) * 49 + tap];
    weff2[tap * 32 + i] = s;
  }
}

// ---------- one-time f16 weight prep: wX[tap][o][i]; Wv, handled natural ----------
__global__ void k_prep(const float* __restrict__ attnW, const float* __restrict__ pos1W,
                       const float* __restrict__ pos2W, const float* __restrict__ Wv,
                       _Float16* __restrict__ wA, _Float16* __restrict__ w1,
                       _Float16* __restrict__ w2, _Float16* __restrict__ wvh) {
  int t = threadIdx.x + blockIdx.x * blockDim.x;  // 28672 total
  if (t < 9216) {
    int tap = t / 1024, r = t % 1024;
    wA[t] = (_Float16)attnW[r * 9 + tap];
  } else if (t < 18432) {
    int q = t - 9216; int tap = q / 1024, r = q % 1024;
    w1[q] = (_Float16)pos1W[r * 9 + tap];
  } else if (t < 27648) {
    int q = t - 18432; int tap = q / 1024, r = q % 1024;
    w2[q] = (_Float16)pos2W[r * 9 + tap];
  } else if (t < 28672) {
    int q = t - 27648;
    wvh[q] = (_Float16)Wv[q];
  }
}

// ---------- spatial mask ----------
__global__ __launch_bounds__(256) void k_spatial_mask(
    const float* __restrict__ xmsi, const float* __restrict__ weff1,
    const float* __restrict__ weff2, const float* __restrict__ c3,
    float* __restrict__ sm, float* __restrict__ out_sm) {
  __shared__ alignas(16) float w1[9 * 32], w2[49 * 32];
  int t = threadIdx.x;
  for (int i = t; i < 9 * 32; i += 256) w1[i] = weff1[i];
  for (int i = t; i < 49 * 32; i += 256) w2[i] = weff2[i];
  __syncthreads();
  int id = blockIdx.x * 256 + t;
  int b = id >> 16;
  int lin = id & 65535;
  int u = lin & 255, v = lin >> 8;  // u = w-index, v = h-index
  const float* base = xmsi + (size_t)b * NN * CC;
  float m1 = 0.f, m2 = 0.f;
  for (int a = 0; a < 3; ++a) {
    int wy = u + a - 1;
    if (wy < 0 || wy > 255) continue;
    for (int b2 = 0; b2 < 3; ++b2) {
      int hx = v + b2 - 1;
      if (hx < 0 || hx > 255) continue;
      const float4* px = (const float4*)(base + ((size_t)hx * WW + wy) * CC);
      const float4* wp = (const float4*)&w1[(a * 3 + b2) * 32];
      for (int i = 0; i < 8; ++i) {
        float4 xv = px[i], wv = wp[i];
        m1 += xv.x * wv.x + xv.y * wv.y + xv.z * wv.z + xv.w * wv.w;
      }
    }
  }
  for (int a = 0; a < 7; ++a) {
    int wy = u + a - 3;
    if (wy < 0 || wy > 255) continue;
    for (int b2 = 0; b2 < 7; ++b2) {
      int hx = v + b2 - 3;
      if (hx < 0 || hx > 255) continue;
      const float4* px = (const float4*)(base + ((size_t)hx * WW + wy) * CC);
      const float4* wp = (const float4*)&w2[(a * 7 + b2) * 32];
      for (int i = 0; i < 8; ++i) {
        float4 xv = px[i], wv = wp[i];
        m2 += xv.x * wv.x + xv.y * wv.y + xv.z * wv.z + xv.w * wv.w;
      }
    }
  }
  float s = 1.f / (1.f + expf(-(c3[0] * m1 + c3[1] * m2)));
  int idx = (b * 256 + u) * 256 + v;  // (b,1,w,h)
  sm[idx] = s;
  out_sm[idx] = s;
}

// ---------- per-batch covariance C = X^T X with WMMA ----------
__global__ __launch_bounds__(256) void k_xtx(const float* __restrict__ xfu,
                                             float* __restrict__ Cmat) {
  __shared__ alignas(32) _Float16 xT[8][1024];  // per-wave transposed chunk [c][pos]
  int t = threadIdx.x;
  int wave = t >> 5, lane = t & 31;
  int wg = blockIdx.x * 8 + wave;  // 2048 waves: 512 per batch
  int b = wg >> 9;
  int w0 = wg & 511;
  const float* base = xfu + (size_t)b * NN * CC;
  v8f a00 = {}, a01 = {}, a10 = {}, a11 = {};
  for (int ci = w0; ci < 2048; ci += 512) {
    __builtin_prefetch(base + ((size_t)ci + 512) * 32 * CC, 0, 1);
    const float4* src = (const float4*)(base + (size_t)ci * 32 * CC);
    _Float16* xl = xT[wave];
#pragma unroll
    for (int q = 0; q < 8; ++q) {  // lane loads position `lane`, 32 channels
      float4 f = src[lane * 8 + q];
      xl[(q * 4 + 0) * 32 + lane] = (_Float16)f.x;
      xl[(q * 4 + 1) * 32 + lane] = (_Float16)f.y;
      xl[(q * 4 + 2) * 32 + lane] = (_Float16)f.z;
      xl[(q * 4 + 3) * 32 + lane] = (_Float16)f.w;
    }
    // wave-private LDS: same-wave DS ops are in order
    v16h A0 = frag_A(xl, lane);            // rows c=0..15
    v16h A1 = frag_A(xl + 16 * 32, lane);  // rows c=16..31
    v16h B0 = frag_B(xl, lane);            // cols c=0..15
    v16h B1 = frag_B(xl + 16 * 32, lane);
    a00 = wmma16(A0, B0, a00);
    a01 = wmma16(A0, B1, a01);
    a10 = wmma16(A1, B0, a10);
    a11 = wmma16(A1, B1, a11);
  }
  int n = lane & 15, hi = lane >> 4;
  float* Cb = Cmat + b * 1024;
#pragma unroll
  for (int r = 0; r < 8; ++r) {
    int m = r + 8 * hi;
    atomicAdd(&Cb[m * 32 + n], a00[r]);
    atomicAdd(&Cb[m * 32 + n + 16], a01[r]);
    atomicAdd(&Cb[(m + 16) * 32 + n], a10[r]);
    atomicAdd(&Cb[(m + 16) * 32 + n + 16], a11[r]);
  }
}

// ---------- attn = softmax(Wk C Wq^T / norms); M = proj x attn, stored f16 ----------
__global__ void k_attn_mat(const float* __restrict__ Cmat, const float* __restrict__ Wq,
                           const float* __restrict__ Wk, const float* __restrict__ projW,
                           _Float16* __restrict__ Mout) {
  __shared__ float Cs[1024], kb[1024], qss[32], attn[4][8][8];
  int b = blockIdx.x;
  int t = threadIdx.x;  // 32 threads
  for (int i = t; i < 1024; i += 32) Cs[i] = Cmat[b * 1024 + i];
  __syncthreads();
  for (int j = 0; j < 32; ++j) {
    float s = 0.f;
    for (int i = 0; i < 32; ++i) s += Wk[t * 32 + i] * Cs[i * 32 + j];
    kb[t * 32 + j] = s;
  }
  float qrow[32];
  for (int j = 0; j < 32; ++j) {
    float s = 0.f;
    for (int i = 0; i < 32; ++i) s += Wq[t * 32 + i] * Cs[i * 32 + j];
    qrow[j] = s;
  }
  float qs = 0.f, ks = 0.f;
  for (int j = 0; j < 32; ++j) {
    qs += qrow[j] * Wq[t * 32 + j];
    ks += kb[t * 32 + j] * Wk[t * 32 + j];
  }
  qss[t] = fmaxf(sqrtf(fmaxf(qs, 0.f)), 1e-12f);
  float kn = fmaxf(sqrtf(fmaxf(ks, 0.f)), 1e-12f);
  __syncthreads();
  int h = t >> 3;
  float g[8];
  for (int e = 0; e < 8; ++e) {
    int col = h * 8 + e;
    float s = 0.f;
    for (int j = 0; j < 32; ++j) s += kb[t * 32 + j] * Wq[col * 32 + j];
    g[e] = s / (kn * qss[col]);
  }
  float mx = g[0];
  for (int e = 1; e < 8; ++e) mx = fmaxf(mx, g[e]);
  float den = 0.f;
  for (int e = 0; e < 8; ++e) { g[e] = expf(g[e] - mx); den += g[e]; }
  for (int e = 0; e < 8; ++e) attn[h][t & 7][e] = g[e] / den;
  __syncthreads();
  for (int hh = 0; hh < 4; ++hh)
    for (int e = 0; e < 8; ++e) {
      float s = 0.f;
      for (int d = 0; d < 8; ++d) s += projW[t * 32 + hh * 8 + d] * attn[hh][d][e];
      Mout[b * 1024 + t * 32 + hh * 8 + e] = (_Float16)s;  // M[o][c_in], natural
    }
}

// ---------- V projection (WMMA) + gating, store (b,w,h,c) f16 ----------
__global__ __launch_bounds__(256) void k_vproj(
    const float* __restrict__ xfu, const _Float16* __restrict__ WvT,
    const float* __restrict__ sm, const float* __restrict__ cm,
    _Float16* __restrict__ vg) {
  __shared__ alignas(32) _Float16 xs[128 * 32];
  __shared__ alignas(32) _Float16 wv[32 * 32];
  __shared__ alignas(16) _Float16 tile[8 * 16 * 32];
  int t = threadIdx.x;
  int p0 = blockIdx.x * 128;  // grid 2048
  int b = p0 >> 16;
  {
    const float4* src = (const float4*)(xfu + (size_t)p0 * 32);
    for (int e = t; e < 1024; e += 256) {
      float4 f = src[e];
      v4h h;
      h[0] = (_Float16)f.x; h[1] = (_Float16)f.y;
      h[2] = (_Float16)f.z; h[3] = (_Float16)f.w;
      *(v4h*)(xs + e * 4) = h;
    }
    const uint4* ws = (const uint4*)WvT;
    uint4* wd = (uint4*)wv;
    for (int e = t; e < 128; e += 256) wd[e] = ws[e];
  }
  __syncthreads();
  int wave = t >> 5, lane = t & 31;
  const _Float16* xl = &xs[wave * 512];
  v16h A = frag_A(xl, lane);
  v8f c0 = {}, c1 = {};
  c0 = wmma16(A, frag_B(wv, lane), c0);
  c1 = wmma16(A, frag_B(wv + 512, lane), c1);
  int n = lane & 15, hi = lane >> 4;
  int pb = (p0 & 65535) + wave * 16;
  const float* smb = sm + b * 65536;
  const float* cmb = cm + b * 32;
#pragma unroll
  for (int r = 0; r < 8; ++r) {
    int m = r + 8 * hi;
    int p = pb + m;
    int hx = p >> 8, wy = p & 255;  // n = hi_idx*W + wi
    float smv = smb[wy * 256 + hx];
    tile[wave * 512 + m * 32 + n] = (_Float16)(c0[r] * smv * cmb[n]);
    tile[wave * 512 + m * 32 + n + 16] = (_Float16)(c1[r] * smv * cmb[n + 16]);
  }
  for (int i = lane; i < 64; i += 32) {  // 16B-chunk stores
    int m = i >> 2, q = i & 3;
    int p = pb + m;
    int hx = p >> 8, wy = p & 255;
    *(uint4*)(vg + ((((size_t)b * 256 + wy) * 256) + hx) * 32 + q * 8) =
        *(const uint4*)(tile + wave * 512 + m * 32 + q * 8);
  }
}

// ---------- 3x3 CxC conv as implicit GEMM with WMMA ----------
// MODE 0: vspec = conv(vg, attn_W) + x_hsi residual            -> dst16
// MODE 1: g     = gelu(conv(vspec, pos1_W) + pos1_b)           -> dst16
// MODE 2: out   = conv(g, pos2_W) + pos2_b
//               + gelu(vspec @ M^T + proj_b)  (final, (b,c,w,h) f32)
template <int MODE>
__global__ __launch_bounds__(256) void k_conv3(
    const _Float16* __restrict__ src, const _Float16* __restrict__ Wf16,
    const float* __restrict__ bias, const float* __restrict__ resid,
    const _Float16* __restrict__ vspec, const _Float16* __restrict__ Mh,
    const float* __restrict__ projb, _Float16* __restrict__ dst16,
    float* __restrict__ dst32) {
  __shared__ alignas(32) _Float16 slab[10 * 18 * 32];            // halo tile
  __shared__ alignas(32) _Float16 wt[9 * 32 * 32];               // [tap][o][i]
  __shared__ alignas(32) _Float16 ctr[MODE == 2 ? 8 * 16 * 32 : 16];
  __shared__ alignas(32) _Float16 mb[MODE == 2 ? 32 * 32 : 16];
  __shared__ alignas(16) float rbuf[MODE == 0 ? 8 * 16 * 32 : 16];
  __shared__ alignas(16) _Float16 ot16[MODE != 2 ? 8 * 16 * 32 : 16];
  __shared__ alignas(16) float ot32[MODE == 2 ? 8 * 16 * 32 : 16];
  int t = threadIdx.x;
  int blk = blockIdx.x;  // grid 2048 = 4 * 32 * 16
  int b = blk >> 9;
  int ut = (blk >> 4) & 31, vt = blk & 15;
  int u0 = ut * 8, v0 = vt * 16;
  const _Float16* sb = src + (size_t)b * NN * CC;
  bool interior = (u0 > 0) && (u0 < 248) && (v0 > 0) && (v0 < 240);
  if (interior) {
    // CDNA5 async global->LDS copy: 10 rows x 1152B contiguous, 16B chunks
    const _Float16* rowbase = sb + (((size_t)(u0 - 1)) * 256 + (v0 - 1)) * 32;
    for (int e = t; e < 720; e += 256) {
      int ur = e / 72, rem = e - ur * 72;
      const void* g = (const void*)((const char*)(rowbase + (size_t)ur * 256 * 32) + rem * 16);
      unsigned loff = (unsigned)(size_t)(&slab[ur * 18 * 32]) + (unsigned)(rem * 16);
      asm volatile("global_load_async_to_lds_b128 %0, %1, off" ::"v"(loff), "v"(g)
                   : "memory");
    }
    asm volatile("s_wait_asynccnt 0" ::: "memory");
  } else {
    for (int e = t; e < 720; e += 256) {
      int ur = e / 72, rem = e - ur * 72;
      int vr = rem >> 2, q = rem & 3;
      int u = u0 + ur - 1, v = v0 + vr - 1;
      v8h val = {};
      if (u >= 0 && u < 256 && v >= 0 && v < 256)
        val = *(const v8h*)(sb + (((size_t)u * 256) + v) * 32 + q * 8);
      *(v8h*)(slab + (ur * 18 + vr) * 32 + q * 8) = val;
    }
  }
  {
    const uint4* wsrc = (const uint4*)Wf16;
    uint4* wdst = (uint4*)wt;
    for (int e = t; e < 1152; e += 256) wdst[e] = wsrc[e];
  }
  if (MODE == 2) {
    const _Float16* vs = vspec + (size_t)b * NN * CC;
    for (int e = t; e < 512; e += 256) {
      int ur = e >> 6, rem = e & 63;
      int vr = rem >> 2, q = rem & 3;
      *(v8h*)(ctr + (ur * 16 + vr) * 32 + q * 8) =
          *(const v8h*)(vs + (((size_t)(u0 + ur)) * 256 + v0 + vr) * 32 + q * 8);
    }
    const uint4* ms = (const uint4*)(Mh + b * 1024);
    uint4* md = (uint4*)mb;
    for (int e = t; e < 128; e += 256) md[e] = ms[e];
  }
  if (MODE == 0) {
    for (int e = t; e < 1024; e += 256) {  // x_hsi residual tile, float4
      int ur = e >> 7, rem = e & 127;
      int pix = rem >> 3, q = rem & 7;
      *(float4*)(rbuf + (ur * 16 + pix) * 32 + q * 4) =
          *(const float4*)(resid + (((size_t)b * 256 + v0 + pix) * 256 + u0 + ur) * 32 + q * 4);
    }
  }
  __syncthreads();
  int wave = t >> 5, lane = t & 31;
  v8f acc0 = {}, acc1 = {};
#pragma unroll
  for (int tap = 0; tap < 9; ++tap) {
    int ky = tap / 3, kx = tap % 3;
    const _Float16* arow = &slab[((wave + ky) * 18 + kx) * 32];
    v16h A = frag_A(arow, lane);
    acc0 = wmma16(A, frag_B(wt + tap * 1024, lane), acc0);
    acc1 = wmma16(A, frag_B(wt + tap * 1024 + 512, lane), acc1);
  }
  int n = lane & 15, hi = lane >> 4;
  int u = u0 + wave;
  if (MODE == 2) {
    v8f t0 = {}, t1 = {};
    v16h A = frag_A(ctr + wave * 512, lane);
    t0 = wmma16(A, frag_B(mb, lane), t0);
    t1 = wmma16(A, frag_B(mb + 512, lane), t1);
#pragma unroll
    for (int r = 0; r < 8; ++r) {
      int m = r + 8 * hi;
      float w0 = gelu_exact(t0[r] + projb[n]);
      float w1 = gelu_exact(t1[r] + projb[n + 16]);
      ot32[wave * 512 + n * 16 + m] = acc0[r] + bias[n] + w0;         // [c][v]
      ot32[wave * 512 + (n + 16) * 16 + m] = acc1[r] + bias[n + 16] + w1;
    }
    for (int i = lane; i < 128; i += 32) {  // per (c): 16 v-floats contiguous
      int c = i >> 2, q = i & 3;
      *(float4*)(dst32 + ((((size_t)b * 32 + c) * 256 + u) * 256) + v0 + q * 4) =
          *(const float4*)(ot32 + wave * 512 + c * 16 + q * 4);
    }
  } else {
#pragma unroll
    for (int r = 0; r < 8; ++r) {
      int m = r + 8 * hi;
      float o0 = acc0[r], o1 = acc1[r];
      if (MODE == 0) {
        o0 += rbuf[wave * 512 + m * 32 + n];
        o1 += rbuf[wave * 512 + m * 32 + n + 16];
      } else {
        o0 = gelu_exact(o0 + bias[n]);
        o1 = gelu_exact(o1 + bias[n + 16]);
      }
      ot16[wave * 512 + m * 32 + n] = (_Float16)o0;
      ot16[wave * 512 + m * 32 + n + 16] = (_Float16)o1;
    }
    for (int i = lane; i < 64; i += 32) {
      int m = i >> 2, q = i & 3;
      *(uint4*)(dst16 + ((((size_t)b * 256 + u) * 256) + v0 + m) * 32 + q * 8) =
          *(const uint4*)(ot16 + wave * 512 + m * 32 + q * 8);
    }
  }
}

extern "C" void kernel_launch(void* const* d_in, const int* in_sizes, int n_in,
                              void* d_out, int out_size, void* d_ws, size_t ws_size,
                              hipStream_t stream) {
  const float* x_fu = (const float*)d_in[0];
  const float* x_msi = (const float*)d_in[1];
  const float* x_hsi = (const float*)d_in[2];
  const float* Wq = (const float*)d_in[3];
  const float* Wk = (const float*)d_in[4];
  const float* Wv = (const float*)d_in[5];
  const float* proj_W = (const float*)d_in[6];
  const float* proj_b = (const float*)d_in[7];
  const float* pos1_W = (const float*)d_in[8];
  const float* pos1_b = (const float*)d_in[9];
  const float* pos2_W = (const float*)d_in[10];
  const float* pos2_b = (const float*)d_in[11];
  const float* ca_fc1 = (const float*)d_in[12];
  const float* ca_fc2 = (const float*)d_in[13];
  const float* sa_c1a = (const float*)d_in[14];
  const float* sa_c1b = (const float*)d_in[15];
  const float* sa_c2a = (const float*)d_in[16];
  const float* sa_c2b = (const float*)d_in[17];
  const float* sa_c3 = (const float*)d_in[18];
  const float* attn_W = (const float*)d_in[19];

  float* out = (float*)d_out;
  float* out_cm = out + (size_t)Bb * CC * NN;  // 8388608
  float* out_sm = out_cm + Bb * CC;            // +128

  char* w = (char*)d_ws;
  float* ws_C = (float*)(w);                       // 16384 B
  float* ws_avg = (float*)(w + 16384);             // 512 B
  float* ws_cm = (float*)(w + 16896);              // 512 B
  float* ws_weff1 = (float*)(w + 17408);           // 1152 B
  float* ws_weff2 = (float*)(w + 18560);           // 6272 B
  _Float16* ws_Mh = (_Float16*)(w + 24832);        // 8192 B
  _Float16* ws_wA = (_Float16*)(w + 33024);        // 18432 B
  _Float16* ws_w1 = (_Float16*)(w + 51456);        // 18432 B
  _Float16* ws_w2 = (_Float16*)(w + 69888);        // 18432 B
  _Float16* ws_wv = (_Float16*)(w + 88320);        // 2048 B
  float* ws_sm = (float*)(w + 90368);              // 1 MB
  _Float16* ws_vg = (_Float16*)(w + 1138944);      // 16 MB
  _Float16* ws_vspec = (_Float16*)(w + 17916160);  // 16 MB
  _Float16* ws_g = (_Float16*)(w + 34693376);      // 16 MB (total ~49 MB)

  hipMemsetAsync(d_ws, 0, 16896, stream);  // zero C + avg accumulators

  k_chan_avg<<<Bb * 64, 256, 0, stream>>>(x_hsi, ws_avg);
  k_chan_mask<<<1, 128, 0, stream>>>(ws_avg, ca_fc1, ca_fc2, ws_cm, out_cm);
  k_weff<<<1, 256, 0, stream>>>(sa_c1a, sa_c1b, sa_c2a, sa_c2b, ws_weff1, ws_weff2);
  k_prep<<<112, 256, 0, stream>>>(attn_W, pos1_W, pos2_W, Wv, ws_wA, ws_w1, ws_w2, ws_wv);
  k_spatial_mask<<<Bb * NN / 256, 256, 0, stream>>>(x_msi, ws_weff1, ws_weff2, sa_c3,
                                                    ws_sm, out_sm);
  k_xtx<<<256, 256, 0, stream>>>(x_fu, ws_C);
  k_attn_mat<<<Bb, 32, 0, stream>>>(ws_C, Wq, Wk, proj_W, ws_Mh);
  k_vproj<<<2048, 256, 0, stream>>>(x_fu, ws_wv, ws_sm, ws_cm, ws_vg);
  k_conv3<0><<<2048, 256, 0, stream>>>(ws_vg, ws_wA, nullptr, x_hsi, nullptr, nullptr,
                                       nullptr, ws_vspec, nullptr);
  k_conv3<1><<<2048, 256, 0, stream>>>(ws_vspec, ws_w1, pos1_b, nullptr, nullptr,
                                       nullptr, nullptr, ws_g, nullptr);
  k_conv3<2><<<2048, 256, 0, stream>>>(ws_g, ws_w2, pos2_b, nullptr, ws_vspec, ws_Mh,
                                       proj_b, nullptr, out);
}
